// Model2_36713380446230
// MI455X (gfx1250) — matrix operand order repaired and checked by reading, compiled, not run on hardware
//
#include <hip/hip_runtime.h>

// ---------------------------------------------------------------------------
// Per-token fused BiLSTM-as-MLP on gfx1250.
//  - GEMMs via v_wmma_f32_16x16x32_bf16 (f-gate skipped: c0==0 => f*c0==0)
//  - Weight tiles staged in LDS per workgroup with the Tensor Data Mover
//    (tensor_load_to_lds + s_wait_tensorcnt), double-buffered over the t loop.
// Workspace: packed bf16 weight fragments, 794,624 bytes.
// ---------------------------------------------------------------------------

typedef __attribute__((ext_vector_type(16))) __bf16 v16bf;
typedef __attribute__((ext_vector_type(8)))  float  v8f;
typedef unsigned int u32x4 __attribute__((ext_vector_type(4)));
typedef int          i32x8 __attribute__((ext_vector_type(8)));
typedef int          i32x4 __attribute__((ext_vector_type(4)));

#define HSZ   128
#define DIN   256          // 2H
#define GATES 512          // 4H
#define NTOK  65536        // B*T = 32*2048
#define TOK_PER_WAVE 16
#define WAVES_PER_BLK 8

// Packed weight tiles, bf16 B-fragment layout [8 kchunk][32 lane][16 elem].
// Gate tiles for one (layer,dir,t) group are CONTIGUOUS (i,g,o) so one 1-D
// TDM descriptor stages all 24 KB:
//   tile = (((lm1*2 + d)*8 + t)*3 + gate)          gate: 0=i, 1=g, 2=o
#define TILE_BF16 4096                     // 8*32*16
#define GROUP_BYTES (3 * TILE_BF16 * 2)    // 24576 B per (l,d,t) group
#define LW_TILES  96
#define LW_TOTAL  (LW_TILES * TILE_BF16)   // 393216 bf16
#define FC_OFF    LW_TOTAL
#define WS_BF16   (LW_TOTAL + TILE_BF16)   // 397312 bf16 = 794,624 bytes

#if __has_builtin(__builtin_amdgcn_tensor_load_to_lds) && \
    __has_builtin(__builtin_amdgcn_s_wait_tensorcnt)
#define HAVE_TDM 1
#else
#define HAVE_TDM 0
#endif

// --------------------------- weight repack kernel ---------------------------
__global__ void prep_pack(const float* __restrict__ w_ih,
                          const float* __restrict__ w_fc,
                          __bf16* __restrict__ wpack) {
    int gid = blockIdx.x * blockDim.x + threadIdx.x;
    if (gid >= WS_BF16) return;
    if (gid < LW_TOTAL) {
        int tile   = gid / TILE_BF16;
        int within = gid % TILE_BF16;
        int c    = within / 512;           // 32-K chunk
        int lane = (within % 512) / 16;    // fragment lane
        int e    = within % 16;            // element within v16bf
        int gate = tile % 3;  int q = tile / 3;
        int t    = q % 8;     q /= 8;
        int d    = q % 2;     int lm1 = q / 2;
        const int gbase = (gate == 0) ? 0 : (gate == 1) ? 256 : 384;
        int n = lane & 15;                               // output column
        int k = c * 32 + ((e < 8) ? ((lane >> 4) * 8 + e) : (16 + (lane >> 4) * 8 + (e - 8)));           // B-layout K
        int j = gbase + t * 16 + n;                      // gate row in [0,512)
        float v = w_ih[((size_t)((lm1 * 2 + d) * GATES + j)) * DIN + k];
        wpack[gid] = (__bf16)v;
    } else {
        int idx  = gid - LW_TOTAL;
        int c    = idx / 512;
        int lane = (idx % 512) / 16;
        int e    = idx % 16;
        int n    = lane & 15;
        int k    = c * 32 + ((e < 8) ? ((lane >> 4) * 8 + e) : (16 + (lane >> 4) * 8 + (e - 8)));
        float v  = (n < 2) ? w_fc[n * DIN + k] : 0.0f;   // zero-padded classes
        wpack[gid] = (__bf16)v;
    }
}

// ------------------------------ device helpers -----------------------------
__device__ __forceinline__ float sigf(float x) { return 1.0f / (1.0f + __expf(-x)); }

// LDS address (bf16 units, inside one wave's 4096-element A region) of the
// A-fragment element for activation channel K, token row m (16-bit A 16x32).
__device__ __forceinline__ int afrag_addr(int K, int m) {
    int c = K >> 5, kl = K & 31, e, half;
    if (kl < 16) { e = kl & 7;               half = kl >> 3; }
    else         { e = 8 + ((kl - 16) & 7);  half = (kl - 16) >> 3; }
    return c * 512 + (half * 16 + m) * 16 + e;
}

#if HAVE_TDM
// Issue a 1-D TDM copy of `bytes` (multiple of 8) from global to LDS.
__device__ __forceinline__ void tdm_stage(unsigned lds_off,
                                          unsigned long long gaddr,
                                          unsigned bytes) {
    unsigned units = bytes >> 3;                       // 8-byte elements
    u32x4 g0;
    g0.x = 1u;                                         // count=1, user mode
    g0.y = lds_off;                                    // LDS byte address
    g0.z = (unsigned)(gaddr & 0xFFFFFFFFull);          // global_addr[31:0]
    g0.w = (unsigned)((gaddr >> 32) & 0x01FFFFFFull)   // global_addr[56:32]
         | (2u << 30);                                 // type = 2 ("image")
    i32x8 g1;
    g1[0] = (3 << 16);                 // wg_mask=0, data_size=8B, no extras
    g1[1] = (int)(units << 16);        // tensor_dim0[15:0]  -> bits [63:48]
    g1[2] = (1 << 16);                 // tensor_dim1 = 1
    g1[3] = (int)(units << 16);        // tile_dim0          -> bits [127:112]
    g1[4] = 0;                         // tile_dim1/2 unused
    g1[5] = (int)units;                // tensor_dim0_stride[31:0]
    g1[6] = 0;
    g1[7] = 0;
    i32x4 z4 = {0, 0, 0, 0};
#if __clang_major__ >= 23
    i32x8 z8 = {0, 0, 0, 0, 0, 0, 0, 0};
    __builtin_amdgcn_tensor_load_to_lds(g0, g1, z4, z4, z8, 0);
#else
    __builtin_amdgcn_tensor_load_to_lds(g0, g1, z4, z4, 0);
#endif
}
#endif

// ------------------------------ main kernel --------------------------------
__global__ __launch_bounds__(256) void lstm_wmma(
    const float* __restrict__ x,
    const float* __restrict__ w_ih0,
    const float* __restrict__ b_ih,
    const float* __restrict__ b_hh,
    const float* __restrict__ b_fc,
    const __bf16* __restrict__ wpack,
    float* __restrict__ out)
{
    __shared__ __bf16 Asmem[WAVES_PER_BLK * 4096];  // 64 KB activation frags
    __shared__ __bf16 Wsmem[2][3 * TILE_BF16];      // 2 x 24 KB weight slots

    const int lane = threadIdx.x & 31;
    const int wv   = threadIdx.x >> 5;
    const int tok0 = (blockIdx.x * WAVES_PER_BLK + wv) * TOK_PER_WAVE;
    __bf16* A_lds = &Asmem[wv * 4096];

    // ---- layer 0: input dim 1, pure elementwise; write A-frags for layer 1
    {
        const int d = lane >> 4;        // half-wave per direction
        const int m = lane & 15;        // token row
        float xv = x[tok0 + m];
        const float* w0 = w_ih0 + d * GATES;
        const float* bi = b_ih  + d * GATES;     // layer 0
        const float* bh = b_hh  + d * GATES;
        for (int hc = 0; hc < HSZ; ++hc) {
            float gi = w0[hc]        * xv + bi[hc]        + bh[hc];
            float gg = w0[256 + hc]  * xv + bi[256 + hc]  + bh[256 + hc];
            float go = w0[384 + hc]  * xv + bi[384 + hc]  + bh[384 + hc];
            float h  = sigf(go) * tanhf(sigf(gi) * tanhf(gg));
            A_lds[afrag_addr(d * HSZ + hc, m)] = (__bf16)h;
        }
    }

    const int n = lane & 15;

    // ---- layers 1..2: [16 x 256] @ [256 x (i,g,o per dir)] via WMMA bf16
    for (int l = 1; l <= 2; ++l) {
        v16bf A[8];
#pragma unroll
        for (int c = 0; c < 8; ++c)
            A[c] = *(const v16bf*)&A_lds[c * 512 + lane * 16];

        const float* bi = b_ih + l * 2 * GATES;
        const float* bh = b_hh + l * 2 * GATES;
        for (int d = 0; d < 2; ++d) {
            const int seg = ((l - 1) * 2 + d) * 8;     // first group of segment
#if HAVE_TDM
            if (wv == 0)                                // prologue: stage t=0
                tdm_stage((unsigned)(size_t)&Wsmem[0][0],
                          (unsigned long long)(size_t)wpack
                              + (unsigned long long)seg * GROUP_BYTES,
                          GROUP_BYTES);
#endif
            for (int t = 0; t < 8; ++t) {
                int slot;
#if HAVE_TDM
                slot = t & 1;
                if (wv == 0) {
                    if (t < 7) {                        // stage t+1, wait t
                        tdm_stage((unsigned)(size_t)&Wsmem[1 - slot][0],
                                  (unsigned long long)(size_t)wpack
                                      + (unsigned long long)(seg + t + 1) * GROUP_BYTES,
                                  GROUP_BYTES);
                        __builtin_amdgcn_s_wait_tensorcnt(1);
                    } else {
                        __builtin_amdgcn_s_wait_tensorcnt(0);
                    }
                }
#else
                slot = 0;
                {   // cooperative fallback copy: 24 KB, 16 B per transfer
                    const uint4* src = (const uint4*)((const char*)wpack
                                       + (size_t)(seg + t) * GROUP_BYTES);
                    uint4* dst = (uint4*)&Wsmem[0][0];
                    for (int i = threadIdx.x; i < GROUP_BYTES / 16; i += 256)
                        dst[i] = src[i];
                }
#endif
                __syncthreads();                        // weights visible to WG

                float bI = bi[d*GATES +   0 + t*16 + n] + bh[d*GATES +   0 + t*16 + n];
                float bG = bi[d*GATES + 256 + t*16 + n] + bh[d*GATES + 256 + t*16 + n];
                float bO = bi[d*GATES + 384 + t*16 + n] + bh[d*GATES + 384 + t*16 + n];
                v8f ai = {bI,bI,bI,bI,bI,bI,bI,bI};
                v8f ag = {bG,bG,bG,bG,bG,bG,bG,bG};
                v8f ao = {bO,bO,bO,bO,bO,bO,bO,bO};

                const __bf16* Wt = &Wsmem[slot][0];
                const v16bf* Bi = (const v16bf*)&Wt[0 * TILE_BF16] + lane;
                const v16bf* Bg = (const v16bf*)&Wt[1 * TILE_BF16] + lane;
                const v16bf* Bo = (const v16bf*)&Wt[2 * TILE_BF16] + lane;
#pragma unroll
                for (int c = 0; c < 8; ++c) {
                    ai = __builtin_amdgcn_wmma_f32_16x16x32_bf16(
                             false, A[c], false, Bi[c * 32], (short)0, ai, false, false);
                    ag = __builtin_amdgcn_wmma_f32_16x16x32_bf16(
                             false, A[c], false, Bg[c * 32], (short)0, ag, false, false);
                    ao = __builtin_amdgcn_wmma_f32_16x16x32_bf16(
                             false, A[c], false, Bo[c * 32], (short)0, ao, false, false);
                }

                // fused gate nonlinearity; store next layer's A fragment
                const int K = d * HSZ + t * 16 + n;
#pragma unroll
                for (int r = 0; r < 8; ++r) {
                    int m = (lane < 16) ? r : (r + 8);
                    float h = sigf(ao[r]) * tanhf(sigf(ai[r]) * tanhf(ag[r]));
                    A_lds[afrag_addr(K, m)] = (__bf16)h;
                }
                __syncthreads();                        // slot reusable
            }
        }
    }

    // ---- final FC: one zero-padded 16-column WMMA tile (classes in n=0,1)
    {
        v16bf A[8];
#pragma unroll
        for (int c = 0; c < 8; ++c)
            A[c] = *(const v16bf*)&A_lds[c * 512 + lane * 16];
        float bb = (n < 2) ? b_fc[n] : 0.0f;
        v8f acc = {bb,bb,bb,bb,bb,bb,bb,bb};
        const v16bf* Bf = (const v16bf*)wpack + (FC_OFF / 16) + lane;
#pragma unroll
        for (int c = 0; c < 8; ++c)
            acc = __builtin_amdgcn_wmma_f32_16x16x32_bf16(
                      false, A[c], false, Bf[c * 32], (short)0, acc, false, false);
        if (n < 2) {
#pragma unroll
            for (int r = 0; r < 8; ++r) {
                int m = (lane < 16) ? r : (r + 8);
                out[(size_t)(tok0 + m) * 2 + n] = acc[r];
            }
        }
    }
}

// ------------------------------ launcher -----------------------------------
extern "C" void kernel_launch(void* const* d_in, const int* in_sizes, int n_in,
                              void* d_out, int out_size, void* d_ws, size_t ws_size,
                              hipStream_t stream) {
    const float* x     = (const float*)d_in[0];
    const float* w_ih0 = (const float*)d_in[1];
    const float* w_ih  = (const float*)d_in[2];
    const float* b_ih  = (const float*)d_in[3];
    const float* b_hh  = (const float*)d_in[4];
    const float* w_fc  = (const float*)d_in[5];
    const float* b_fc  = (const float*)d_in[6];
    __bf16* wpack = (__bf16*)d_ws;     // needs WS_BF16*2 = 794,624 bytes
    float* out = (float*)d_out;

    prep_pack<<<(WS_BF16 + 255) / 256, 256, 0, stream>>>(w_ih, w_fc, wpack);

    const int blocks = NTOK / (TOK_PER_WAVE * WAVES_PER_BLK);   // 512
    lstm_wmma<<<blocks, 256, 0, stream>>>(x, w_ih0, b_ih, b_hh, b_fc, wpack, out);
}
